// Sequence_Decoder_V1_69441031242045
// MI455X (gfx1250) — matrix operand order repaired and checked by reading, not compiled, not run on hardware
//
#include <hip/hip_runtime.h>
#include <hip/hip_bf16.h>

// ---------------------------------------------------------------------------
// Sequence decoder (LSTM, T=512) for MI455X / gfx1250, wave32 + WMMA bf16.
//
// Math rewrite: x_t = h_{t-1} + x0  =>  gates_t = h_{t-1} @ (W_ih+W_hh)^T + X0proj
// where X0proj = x0 @ W_ih^T + b_ih + b_hh (constant across steps).
//
// The recurrence is independent per batch row, so partition over batch only:
// 16 WGs x 512 threads; each WG owns 16 batch rows for all 512 steps.
// h lives in LDS (ping-pong), c in VGPRs, W_comb in registers (i,f) + LDS (g,o).
// No global synchronization: one __syncthreads per step.
// ---------------------------------------------------------------------------

typedef __attribute__((ext_vector_type(16))) __bf16       v16bf;
typedef __attribute__((ext_vector_type(8)))  float        v8f;
typedef __attribute__((ext_vector_type(4)))  unsigned int u32x4;
typedef unsigned short u16b;

#define BATCH   256
#define HID     256
#define TSTEPS  512
#define NOUT    702
#define NOUTP   704            // padded to 16
#define LROW    264            // padded LDS row stride (elements) for bank spread

union ABu {
    v16bf v;
    __bf16 h[16];
    u32x4  q[2];
};

__device__ __forceinline__ float sigf(float x)   { return 1.0f / (1.0f + __expf(-x)); }
__device__ __forceinline__ float tanhfast(float x){ return 2.0f / (1.0f + __expf(-2.0f * x)) - 1.0f; }
__device__ __forceinline__ u16b f2bf(float x) {
    __bf16 b = (__bf16)x; u16b u; __builtin_memcpy(&u, &b, 2); return u;
}

// ---- prep: W_comb = bf16(W_ih + W_hh), row-major [1024][256] ----
__global__ void prep_w_kernel(const float* __restrict__ Wih,
                              const float* __restrict__ Whh,
                              u16b* __restrict__ Wc) {
    int i = blockIdx.x * 256 + threadIdx.x;       // 0..262143
    Wc[i] = f2bf(Wih[i] + Whh[i]);
}

// ---- prep: W_rec bf16, padded to [704][256] ----
__global__ void prep_wrec_kernel(const float* __restrict__ Wrec,
                                 u16b* __restrict__ Wr) {
    int n = blockIdx.x, k = threadIdx.x;
    float v = (n < NOUT) ? Wrec[n * 256 + k] : 0.0f;
    Wr[n * 256 + k] = f2bf(v);
}

// ---- prep: X0proj[m][n] = b_ih[n] + b_hh[n] + sum_k x0[m][k]*W_ih[n][k] (fp32) ----
__global__ void x0proj_kernel(const float* __restrict__ x0,
                              const float* __restrict__ Wih,
                              const float* __restrict__ bih,
                              const float* __restrict__ bhh,
                              float* __restrict__ X0p) {
    int m = blockIdx.x;
    int n = blockIdx.y * 256 + threadIdx.x;
    const float* xr = x0 + m * 256;
    const float* wr = Wih + n * 256;
    float s = bih[n] + bhh[n];
#pragma unroll 8
    for (int k = 0; k < 256; ++k) s = fmaf(xr[k], wr[k], s);
    X0p[m * 1024 + n] = s;
}

// ---- recurrent kernel: 16 WGs x 512 threads (16 waves), batch-partitioned ----
__global__ __launch_bounds__(512) void lstm_kernel(
    const float* __restrict__ X0p,     // [256][1024]
    const u16b* __restrict__ Wc,       // bf16 [1024][256]
    float* __restrict__ dec) {         // fp32 decoded (B,T,H) region of d_out

    // B operand for gates {g,o}: 32 tiles of 16 rows, padded row stride.
    __shared__ __align__(16) u16b sB[512 * LROW];        // 512 rows  (~264 KB)
    __shared__ __align__(16) u16b sH[2][16 * LROW];      // h ping-pong (~16.5 KB)

    const int tid  = threadIdx.x;
    const int m0   = blockIdx.x * 16;              // batch rows owned by this WG
    const int wave = tid >> 5, lane = tid & 31;
    const int j0   = wave * 16;                    // hidden tile owned by this wave
    const int nl   = lane & 15;
    const int hi8  = (lane >= 16) ? 8 : 0;
    const int kb_a = (lane < 16) ? 0 : 16;         // A-frag K phase
    const int kb_b = (lane < 16) ? 0 : 16;         // B-frag K phase

    // ---- preload gate {g,o} weights into LDS ----
    // local row r = (w*2+tt)*16+nlr  <->  global gate row (2+tt)*256 + w*16 + nlr
    for (int idx = tid; idx < 512 * 32; idx += 512) {
        int row = idx >> 5;                  // 0..511
        int off = (idx & 31) * 8;            // 0..248 (k offset, 8 elems)
        int w   = row >> 5;
        int tt  = (row >> 4) & 1;
        int nlr = row & 15;
        int grow = (2 + tt) * 256 + w * 16 + nlr;
        *(u32x4*)&sB[row * LROW + off] = *(const u32x4*)&Wc[grow * 256 + off];
    }

    // ---- B for gates {i,f} in registers (128 VGPRs) ----
    ABu bI[8], bF[8];
    {
        const u16b* rI = Wc + (size_t)(0 * 256 + j0 + nl) * 256 + kb_b;
        const u16b* rF = Wc + (size_t)(1 * 256 + j0 + nl) * 256 + kb_b;
#pragma unroll
        for (int kk = 0; kk < 8; ++kk) {
            bI[kk].q[0] = *(const u32x4*)&rI[kk * 32];
            bI[kk].q[1] = *(const u32x4*)&rI[kk * 32 + 8];
            bF[kk].q[0] = *(const u32x4*)&rF[kk * 32];
            bF[kk].q[1] = *(const u32x4*)&rF[kk * 32 + 8];
        }
    }

    // ---- X0proj accumulator-init in registers (4 gates x v8f) ----
    // C-layout element (v,lane): m = m0 + v + hi8, n = G*256 + j0 + nl
    v8f x0c[4];
#pragma unroll
    for (int G = 0; G < 4; ++G) {
        const float* p = X0p + (size_t)(m0 + hi8) * 1024 + G * 256 + j0 + nl;
#pragma unroll
        for (int v = 0; v < 8; ++v) x0c[G][v] = p[v * 1024];
    }

    // Cell state: lane owns (m = v + hi8, j = j0 + nl) for v = 0..7
    float creg[8];
#pragma unroll
    for (int v = 0; v < 8; ++v) creg[v] = 0.f;

    const int sBg = ((wave * 2 + 0) * 16 + nl) * LROW;   // gate g tile row base
    const int sBo = ((wave * 2 + 1) * 16 + nl) * LROW;   // gate o tile row base

    __syncthreads();   // sB ready

    for (int t = 0; t < TSTEPS; ++t) {
        v8f accI = x0c[0], accF = x0c[1], accG = x0c[2], accO = x0c[3];

        if (t > 0) {
            const u16b* hprev = sH[(t - 1) & 1];
#pragma unroll
            for (int kk = 0; kk < 8; ++kk) {
                const int k0 = kk * 32;
                ABu a, bg, bo;
                a.q[0]  = *(const u32x4*)&hprev[nl * LROW + k0 + kb_a];
                a.q[1]  = *(const u32x4*)&hprev[nl * LROW + k0 + kb_a + 8];
                bg.q[0] = *(const u32x4*)&sB[sBg + k0 + kb_b];
                bg.q[1] = *(const u32x4*)&sB[sBg + k0 + kb_b + 8];
                bo.q[0] = *(const u32x4*)&sB[sBo + k0 + kb_b];
                bo.q[1] = *(const u32x4*)&sB[sBo + k0 + kb_b + 8];
                accI = __builtin_amdgcn_wmma_f32_16x16x32_bf16(
                    false, a.v, false, bI[kk].v, (short)0, accI, false, false);
                accF = __builtin_amdgcn_wmma_f32_16x16x32_bf16(
                    false, a.v, false, bF[kk].v, (short)0, accF, false, false);
                accG = __builtin_amdgcn_wmma_f32_16x16x32_bf16(
                    false, a.v, false, bg.v, (short)0, accG, false, false);
                accO = __builtin_amdgcn_wmma_f32_16x16x32_bf16(
                    false, a.v, false, bo.v, (short)0, accO, false, false);
            }
        }

        // LSTM cell entirely in-lane: the 4 accumulators are element-aligned.
        u16b* hnext = (u16b*)sH[t & 1];
#pragma unroll
        for (int v = 0; v < 8; ++v) {
            float cn = sigf(accF[v]) * creg[v] + sigf(accI[v]) * tanhfast(accG[v]);
            creg[v]  = cn;
            float hv = sigf(accO[v]) * tanhfast(cn);
            int m = v + hi8;
            hnext[m * LROW + j0 + nl] = f2bf(hv);
            dec[((size_t)(m0 + m) * TSTEPS + t) * HID + (j0 + nl)] = hv;
        }

        __syncthreads();   // step t writes visible before step t+1 reads
    }
}

// ---- projection: recon[r][n] = b_rec[n] + sum_k dec[r][k]*W_rec[n][k] ----
__global__ __launch_bounds__(256) void proj_kernel(
    const float* __restrict__ dec,     // (B*T, 256) fp32
    const u16b* __restrict__ Wr,       // bf16 [704][256]
    const float* __restrict__ brec,    // [702]
    float* __restrict__ out) {         // (B*T, 702)

    const int wave = threadIdx.x >> 5, lane = threadIdx.x & 31;
    const int mt = blockIdx.x * 8 + wave;                 // 0..8191 (16-row M tiles)

    // Cache the full 16x256 A strip as 8 bf16 fragments in registers.
    const float* arow = dec + (size_t)(mt * 16 + (lane & 15)) * 256;
    const int kb = (lane < 16) ? 0 : 8;
    ABu af[8];
#pragma unroll
    for (int f = 0; f < 8; ++f) {
        int k0 = f * 32 + kb;
#pragma unroll
        for (int i = 0; i < 8; ++i) af[f].h[i]     = (__bf16)arow[k0 + i];
#pragma unroll
        for (int i = 0; i < 8; ++i) af[f].h[8 + i] = (__bf16)arow[k0 + 16 + i];
    }

    const int kbB = (lane < 16) ? 0 : 8;
    for (int nt = 0; nt < NOUTP / 16; ++nt) {
        int n = nt * 16 + (lane & 15);
        float bv = (n < NOUT) ? brec[n] : 0.0f;
        v8f acc = {bv, bv, bv, bv, bv, bv, bv, bv};
        const u16b* bbase = Wr + (size_t)n * 256 + kbB;
#pragma unroll
        for (int k0 = 0; k0 < 256; k0 += 32) {
            ABu b;
            b.q[0] = *(const u32x4*)&bbase[k0];
            b.q[1] = *(const u32x4*)&bbase[k0 + 16];
            acc = __builtin_amdgcn_wmma_f32_16x16x32_bf16(
                false, af[k0 >> 5].v, false, b.v, (short)0, acc, false, false);
        }
        if (n < NOUT) {
            int rbase = mt * 16 + ((lane >= 16) ? 8 : 0);
#pragma unroll
            for (int v = 0; v < 8; ++v)
                out[(size_t)(rbase + v) * NOUT + n] = acc[v];
        }
    }
}

extern "C" void kernel_launch(void* const* d_in, const int* in_sizes, int n_in,
                              void* d_out, int out_size, void* d_ws, size_t ws_size,
                              hipStream_t stream) {
    const float* embed = (const float*)d_in[0];   // (256,1,256)
    const float* W_ih  = (const float*)d_in[1];   // (1024,256)
    const float* W_hh  = (const float*)d_in[2];   // (1024,256)
    const float* b_ih  = (const float*)d_in[3];   // (1024)
    const float* b_hh  = (const float*)d_in[4];   // (1024)
    const float* W_rec = (const float*)d_in[5];   // (702,256)
    const float* b_rec = (const float*)d_in[6];   // (702)
    (void)in_sizes; (void)n_in; (void)out_size; (void)ws_size;

    char* ws = (char*)d_ws;
    float* X0p = (float*)ws;                      // 1 MB
    u16b*  Wc  = (u16b*)(ws + 1048576);           // 512 KB
    u16b*  Wr  = (u16b*)(ws + 1572864);           // 352 KB

    float* recon = (float*)d_out;                                  // (B,T,702)
    float* dec   = (float*)d_out + (size_t)BATCH * TSTEPS * NOUT;  // (B,T,256)

    prep_w_kernel<<<1024, 256, 0, stream>>>(W_ih, W_hh, Wc);
    prep_wrec_kernel<<<NOUTP, 256, 0, stream>>>(W_rec, Wr);
    x0proj_kernel<<<dim3(256, 4), 256, 0, stream>>>(embed, W_ih, b_ih, b_hh, X0p);
    lstm_kernel<<<16, 512, 0, stream>>>(X0p, Wc, dec);
    proj_kernel<<<1024, 256, 0, stream>>>(dec, Wr, b_rec, recon);
}
